// ShapeletAttentionHead_23158463660291
// MI455X (gfx1250) — matrix operand order repaired and checked by reading, not compiled, not run on hardware
//
#include <hip/hip_runtime.h>

// Problem constants (ShapeletAttentionHead)
#define BB 32
#define NN 1024
#define DD 768
#define KK 64
#define LL 16
#define TT 1009      // N - L + 1
#define TP 1024      // padded T stride for the scores scratch (float4-aligned stores)
#define LDT 776      // LDS token row stride in bf16 elems (768 + 8 pad -> avoids bank conflicts)
#define TROWS 79     // token rows staged per block: 64 (t tile) + 15 (window overhang)

typedef __bf16 bf16;
typedef bf16  v16bf __attribute__((ext_vector_type(16)));
typedef float v8f   __attribute__((ext_vector_type(8)));

union FragB16 { v16bf v; uint4 q[2]; };

// ---------------------------------------------------------------------------
// Kernel 0: one-time shapelet transpose + bf16 cast:  Wt[l][k][d] = sh[k][d][l]
// Makes every WMMA B-fragment two contiguous 16B per-lane reads (L2 resident).
// ---------------------------------------------------------------------------
__global__ __launch_bounds__(256)
void k_shapelet_tr(const float* __restrict__ sh, bf16* __restrict__ wt) {
  int i = blockIdx.x * 256 + threadIdx.x;          // i = k*DD + d
  if (i >= KK * DD) return;
  const float* p = sh + (size_t)i * LL;            // shapelets[k][d][0..15], contiguous
  #pragma unroll
  for (int l = 0; l < LL; ++l)
    wt[(size_t)l * (KK * DD) + i] = (bf16)p[l];
}

// ---------------------------------------------------------------------------
// Kernel 1: scores[b,k,t] = sum_{l,d} tokens[b,t+l,d] * sh[k,d,l]  via WMMA.
// Block = (b, 64-wide t tile). 8 waves = 4 k-tiles x 2 t-subtile PAIRS:
// each wave keeps TWO 16x16 accumulators so every B fragment (2 global B128)
// feeds 2 WMMAs -> halves L2 traffic vs. one-accumulator version.
// A (16x32 bf16) = token windows from LDS, B (32x16 bf16) = Wt from global/L2.
// D tiles are scores^T[t,k]: lane rows are contiguous in t -> float4 stores.
// ---------------------------------------------------------------------------
__global__ __launch_bounds__(256)
void k_scores(const float* __restrict__ tokens, const bf16* __restrict__ wt,
              float* __restrict__ scores) {
  extern __shared__ char smem[];
  bf16* tok = (bf16*)smem;                         // [TROWS][LDT] bf16 (~120 KB)
  const int b   = blockIdx.y;
  const int t0  = blockIdx.x * 64;
  const int tid = threadIdx.x;

  // Stage TROWS token rows (t0 .. t0+78) into LDS as bf16 (f32 -> bf16 cast).
  for (int g = tid; g < TROWS * (DD / 4); g += 256) {
    int r = g / (DD / 4);
    int c = (g % (DD / 4)) * 4;
    int row = t0 + r; if (row > NN - 1) row = NN - 1;   // clamp; only affects t >= T'
    float4 f = *(const float4*)(tokens + ((size_t)b * NN + row) * DD + c);
    bf16* d = tok + r * LDT + c;
    d[0] = (bf16)f.x; d[1] = (bf16)f.y; d[2] = (bf16)f.z; d[3] = (bf16)f.w;
  }
  __syncthreads();

  const int w   = tid >> 5;        // wave id (wave32)
  const int ln  = tid & 31;
  const int kt  = w & 3;           // 16-wide k tile
  const int tp0 = (w >> 2) * 2;    // first of this wave's two 16-wide t subtiles
  const int h   = ln >> 4;         // lane half
  const int nl  = ln & 15;

  v8f acc0 = {0.f, 0.f, 0.f, 0.f, 0.f, 0.f, 0.f, 0.f};
  v8f acc1 = {0.f, 0.f, 0.f, 0.f, 0.f, 0.f, 0.f, 0.f};

  // A element j (16-bit A 16x32 layout): contraction K = (j>=8?16:0) + 8h + (j&7)
  //   -> two contiguous 16B chunks per lane at row (ts*16 + m + l), cols d0+8h(+16)
  const bf16* a0base = tok + (tp0 * 16 + nl) * LDT + h * 8;
  const bf16* a1base = a0base + 16 * LDT;
  // B element j (16-bit B 32x16 layout): row = (j>=8?16:0) + 8h + (j&7), col = nl
  //   -> two contiguous 16B chunks per lane at Wt[l][kt*16+nl][d0 + 8h(+16) .. +7]
  const bf16* bbase = wt + (size_t)(kt * 16 + nl) * DD + h * 8;

  for (int l = 0; l < LL; ++l) {
    const bf16* a0 = a0base + l * LDT;
    const bf16* a1 = a1base + l * LDT;
    const bf16* br = bbase + (size_t)l * (KK * DD);
    #pragma unroll
    for (int dc = 0; dc < DD / 32; ++dc) {         // 24 chunks of 32 along d
      FragB16 bm, x, y;
      bm.q[0] = *(const uint4*)(br + dc * 32);
      bm.q[1] = *(const uint4*)(br + dc * 32 + 16);
      x.q[0]  = *(const uint4*)(a0 + dc * 32);
      x.q[1]  = *(const uint4*)(a0 + dc * 32 + 16);
      y.q[0]  = *(const uint4*)(a1 + dc * 32);
      y.q[1]  = *(const uint4*)(a1 + dc * 32 + 16);
      acc0 = __builtin_amdgcn_wmma_f32_16x16x32_bf16(false, x.v, false, bm.v,
                                                     (short)0, acc0, false, false);
      acc1 = __builtin_amdgcn_wmma_f32_16x16x32_bf16(false, y.v, false, bm.v,
                                                     (short)0, acc1, false, false);
    }
  }

  // D layout: VGPR r, lane (h,nl) holds scores^T[t_base + 8h + r][kt*16+nl]
  const int kidx = kt * 16 + nl;
  const int tb   = t0 + tp0 * 16 + 8 * h;          // 8 consecutive t per lane
  float* p = scores + (((size_t)b * KK + kidx) << 10) + tb;   // stride TP=1024
  *(float4*)(p)      = ((const float4*)&acc0)[0];
  *(float4*)(p + 4)  = ((const float4*)&acc0)[1];
  *(float4*)(p + 16) = ((const float4*)&acc1)[0];
  *(float4*)(p + 20) = ((const float4*)&acc1)[1];
}

// ---------------------------------------------------------------------------
// Kernel 2: row softmax over t (1009 valid of 1024-padded), writes attn output.
// ---------------------------------------------------------------------------
__global__ __launch_bounds__(256)
void k_softmax(const float* __restrict__ scores, float* __restrict__ attn) {
  const int row = blockIdx.x;                      // b*KK + k
  const float* s = scores + ((size_t)row << 10);
  float* a = attn + (size_t)row * TT;
  const float scale = rsqrtf((float)(DD * LL));
  __shared__ float red[256];
  const int tid = threadIdx.x;

  float v[4];
  float m = -3.402823e38f;
  #pragma unroll
  for (int i = 0; i < 4; ++i) {
    int t = tid + i * 256;
    v[i] = (t < TT) ? s[t] * scale : -3.402823e38f;
    m = fmaxf(m, v[i]);
  }
  red[tid] = m; __syncthreads();
  for (int o = 128; o > 0; o >>= 1) { if (tid < o) red[tid] = fmaxf(red[tid], red[tid + o]); __syncthreads(); }
  m = red[0]; __syncthreads();

  float e[4], sum = 0.f;
  #pragma unroll
  for (int i = 0; i < 4; ++i) {
    int t = tid + i * 256;
    e[i] = (t < TT) ? __expf(v[i] - m) : 0.f;
    sum += e[i];
  }
  red[tid] = sum; __syncthreads();
  for (int o = 128; o > 0; o >>= 1) { if (tid < o) red[tid] += red[tid + o]; __syncthreads(); }
  const float inv = 1.f / red[0];
  #pragma unroll
  for (int i = 0; i < 4; ++i) {
    int t = tid + i * 256;
    if (t < TT) a[t] = e[i] * inv;
  }
}

// ---------------------------------------------------------------------------
// Kernel 3: abar[b,t] = mean_k attn[b,k,t]   (k-mean commutes with the einsum)
// ---------------------------------------------------------------------------
__global__ __launch_bounds__(256)
void k_abar(const float* __restrict__ attn, float* __restrict__ abar) {
  const int b = blockIdx.x;
  const int t = blockIdx.y * 256 + threadIdx.x;
  float sum = 0.f;
  if (t < TT) {
    const float* a = attn + (size_t)b * KK * TT + t;
    #pragma unroll 8
    for (int k = 0; k < KK; ++k) sum += a[(size_t)k * TT];
    sum *= (1.f / KK);
  }
  abar[b * TP + t] = sum;
}

// ---------------------------------------------------------------------------
// Kernel 4: w[b,n] = (1/L) * sum_{l<L, 0<=n-l<T'} abar[b, n-l]
//  (window-mean folded back onto token weights)
// ---------------------------------------------------------------------------
__global__ __launch_bounds__(256)
void k_wsum(const float* __restrict__ abar, float* __restrict__ wv) {
  const int b = blockIdx.x;
  const int n = blockIdx.y * 256 + threadIdx.x;
  float s = 0.f;
  #pragma unroll
  for (int l = 0; l < LL; ++l) {
    int t = n - l;
    if (t >= 0 && t < TT) s += abar[b * TP + t];
  }
  wv[b * TP + n] = s * (1.f / LL);
}

// ---------------------------------------------------------------------------
// Kernel 5: rep_agg[b,d] = sum_n w[b,n]*tokens[b,n,d], fused layer-norm over d.
// One block per b; 256 threads each own 3 d-lanes -> full row in registers.
// ---------------------------------------------------------------------------
__global__ __launch_bounds__(256)
void k_rep_ln(const float* __restrict__ tokens, const float* __restrict__ wv,
              float* __restrict__ out) {
  const int b = blockIdx.x;
  const int tid = threadIdx.x;
  __shared__ float wl[NN];
  __shared__ float red[256];
  for (int n = tid; n < NN; n += 256) wl[n] = wv[b * TP + n];
  __syncthreads();

  float x0 = 0.f, x1 = 0.f, x2 = 0.f;
  const float* tp = tokens + (size_t)b * NN * DD;
  for (int n = 0; n < NN; ++n) {
    const float wn = wl[n];
    const float* r = tp + (size_t)n * DD;
    x0 = fmaf(wn, r[tid      ], x0);
    x1 = fmaf(wn, r[tid + 256], x1);
    x2 = fmaf(wn, r[tid + 512], x2);
  }

  float s = x0 + x1 + x2;
  red[tid] = s; __syncthreads();
  for (int o = 128; o > 0; o >>= 1) { if (tid < o) red[tid] += red[tid + o]; __syncthreads(); }
  const float mean = red[0] * (1.f / DD);
  __syncthreads();
  float sq = x0 * x0 + x1 * x1 + x2 * x2;
  red[tid] = sq; __syncthreads();
  for (int o = 128; o > 0; o >>= 1) { if (tid < o) red[tid] += red[tid + o]; __syncthreads(); }
  const float var = red[0] * (1.f / DD) - mean * mean;   // population variance (matches jnp.var)
  const float rs  = rsqrtf(var + 1e-5f);

  float* po = out + (size_t)b * DD;
  po[tid      ] = (x0 - mean) * rs;
  po[tid + 256] = (x1 - mean) * rs;
  po[tid + 512] = (x2 - mean) * rs;
}

// ---------------------------------------------------------------------------
// Launcher. d_out = [rep_agg (B*D f32) | attn (B*K*T' f32)].
// Workspace layout (bytes):
//   [0, 8 MiB)           scores scratch  B*K*TP f32
//   [+8MiB, +128KiB)     abar            B*TP f32
//   [+...,  +128KiB)     w               B*TP f32
//   [+...,  +1.5MiB)     Wt (bf16)       L*K*D
// Total ~10.22 MB.
// ---------------------------------------------------------------------------
extern "C" void kernel_launch(void* const* d_in, const int* in_sizes, int n_in,
                              void* d_out, int out_size, void* d_ws, size_t ws_size,
                              hipStream_t stream) {
  const float* tokens    = (const float*)d_in[0];
  const float* shapelets = (const float*)d_in[1];

  float* out      = (float*)d_out;
  float* rep_out  = out;
  float* attn_out = out + (size_t)BB * DD;

  char* ws = (char*)d_ws;
  const size_t SC_BYTES = (size_t)BB * KK * TP * 4;   // 8,388,608
  const size_t AB_BYTES = (size_t)BB * TP * 4;        // 131,072
  float* ws_scores = (float*)ws;
  float* ws_abar   = (float*)(ws + SC_BYTES);
  float* ws_w      = (float*)(ws + SC_BYTES + AB_BYTES);
  bf16*  ws_wt     = (bf16*)(ws + SC_BYTES + 2 * AB_BYTES);

  k_shapelet_tr<<<dim3((KK * DD + 255) / 256), 256, 0, stream>>>(shapelets, ws_wt);
  k_scores<<<dim3(16, BB), 256, TROWS * LDT * 2, stream>>>(tokens, ws_wt, ws_scores);
  k_softmax<<<dim3(BB * KK), 256, 0, stream>>>(ws_scores, attn_out);
  k_abar<<<dim3(BB, 4), 256, 0, stream>>>(attn_out, ws_abar);
  k_wsum<<<dim3(BB, 4), 256, 0, stream>>>(ws_abar, ws_w);
  k_rep_ln<<<dim3(BB), 256, 0, stream>>>(tokens, ws_w, rep_out);
}
